// EnsembleRSSM_13348758356021
// MI455X (gfx1250) — matrix operand order repaired and checked by reading, not compiled, not run on hardware
//
#include <hip/hip_runtime.h>

typedef __bf16 bf16_t;
typedef __attribute__((ext_vector_type(8)))  __bf16 bf16x8;
typedef __attribute__((ext_vector_type(16))) __bf16 bf16x16;
typedef __attribute__((ext_vector_type(8)))  float  f32x8;

union Frag { bf16x16 v; bf16x8 h[2]; };

// CDNA5 async global->LDS copy (ASYNCcnt). vdst = LDS byte address, vaddr = 64-bit VA.
__device__ __forceinline__ void async_load_b128(uint32_t lds_addr, const void* gptr) {
  asm volatile("global_load_async_to_lds_b128 %0, %1, off"
               :: "v"(lds_addr), "v"((unsigned long long)(uintptr_t)gptr)
               : "memory");
}
__device__ __forceinline__ void wait_async0() {
  asm volatile("s_wait_asynccnt 0x0" ::: "memory");
}
__device__ __forceinline__ uint32_t lds_addr_of(const void* p) {
  return (uint32_t)(uintptr_t)p;  // flat LDS address carries LDS offset in [31:0]
}

// ---------------------------------------------------------------------------
// GEMM: C[M,N] = A[M,K](bf16, row-major) * Wt[N,K](bf16, row-major => B^T) + bias[N]
// Block tile 128x128, K-step 32, 256 threads (8 waves), wave tile 64x32.
// Double-buffered LDS fed by global_load_async_to_lds_b128.
// ---------------------------------------------------------------------------
__global__ __launch_bounds__(256) void gemm_bf16_wmma(
    const bf16_t* __restrict__ A, int lda,
    const bf16_t* __restrict__ Wt, int ldw,
    const float* __restrict__ bias,
    float* __restrict__ C, int ldc, int K)
{
  constexpr int BM = 128, BN = 128, BK = 32, LS = 40; // 40 bf16 = 80B padded row
  __shared__ bf16_t sA[2][BM * LS];
  __shared__ bf16_t sB[2][BN * LS];

  const int tid  = threadIdx.x;
  const int lane = tid & 31;
  const int wave = tid >> 5;
  const int r    = lane & 15;   // row/col within 16-wide fragment
  const int hb   = lane >> 4;   // K-half selector

  const int m0 = blockIdx.y * BM;
  const int n0 = blockIdx.x * BN;
  const int wm = (wave & 1) * 64;   // wave row offset in block
  const int wn = (wave >> 1) * 32;  // wave col offset in block

  f32x8 acc[4][2];
  #pragma unroll
  for (int m = 0; m < 4; ++m)
    #pragma unroll
    for (int n = 0; n < 2; ++n)
      #pragma unroll
      for (int j = 0; j < 8; ++j) acc[m][n][j] = 0.0f;

  // Staging: 256 threads, each owns one row-half: row = tid>>1, 32B at col (tid&1)*16.
  const int srow = tid >> 1;          // 0..127
  const int scol = (tid & 1) * 16;    // 0 or 16
  const bf16_t* gA = A  + (size_t)(m0 + srow) * lda + scol;
  const bf16_t* gB = Wt + (size_t)(n0 + srow) * ldw + scol;
  const int sOff = srow * LS + scol;

  // Prologue: kick tile 0 into buffer 0.
  async_load_b128(lds_addr_of(&sA[0][sOff]),     gA);
  async_load_b128(lds_addr_of(&sA[0][sOff + 8]), gA + 8);
  async_load_b128(lds_addr_of(&sB[0][sOff]),     gB);
  async_load_b128(lds_addr_of(&sB[0][sOff + 8]), gB + 8);

  int buf = 0;
  for (int k0 = 0; k0 < K; k0 += BK, buf ^= 1) {
    wait_async0();        // my share of tile k resident in LDS
    __syncthreads();      // everyone's share resident; alternate buffer free
    if (k0 + BK < K) {    // overlap: tile k+1 transfer vs tile k compute
      const bf16_t* nA = gA + k0 + BK;
      const bf16_t* nB = gB + k0 + BK;
      async_load_b128(lds_addr_of(&sA[buf ^ 1][sOff]),     nA);
      async_load_b128(lds_addr_of(&sA[buf ^ 1][sOff + 8]), nA + 8);
      async_load_b128(lds_addr_of(&sB[buf ^ 1][sOff]),     nB);
      async_load_b128(lds_addr_of(&sB[buf ^ 1][sOff + 8]), nB + 8);
    }

    // B fragments (32x16 each): lane r holds column, 16 contiguous K at hb*16.
    Frag bfr[2];
    #pragma unroll
    for (int n = 0; n < 2; ++n) {
      const int bro = (wn + n * 16 + r) * LS + hb * 16;
      bfr[n].h[0] = *(const bf16x8*)&sB[buf][bro];
      bfr[n].h[1] = *(const bf16x8*)&sB[buf][bro + 8];
    }
    #pragma unroll
    for (int m = 0; m < 4; ++m) {
      // A fragment (16x32): lane r holds row, 16 contiguous K at hb*16.
      Frag af;
      const int aro = (wm + m * 16 + r) * LS;
      af.h[0] = *(const bf16x8*)&sA[buf][aro + hb * 16];
      af.h[1] = *(const bf16x8*)&sA[buf][aro + hb * 16 + 8];
      acc[m][0] = __builtin_amdgcn_wmma_f32_16x16x32_bf16(
          false, af.v, false, bfr[0].v, (short)0, acc[m][0], false, false);
      acc[m][1] = __builtin_amdgcn_wmma_f32_16x16x32_bf16(
          false, af.v, false, bfr[1].v, (short)0, acc[m][1], false, false);
    }
  }

  // Epilogue: C/D layout — VGPR v holds (M = v + 8*hb, N = r).
  #pragma unroll
  for (int n = 0; n < 2; ++n) {
    const int col = n0 + wn + n * 16 + r;
    const float bv = bias[col];
    #pragma unroll
    for (int m = 0; m < 4; ++m) {
      const int rowb = m0 + wm + m * 16 + hb * 8;
      #pragma unroll
      for (int v = 0; v < 8; ++v)
        C[(size_t)(rowb + v) * ldc + col] = acc[m][n][v] + bv;
    }
  }
}

// ---------------------------------------------------------------------------
// Helpers
// ---------------------------------------------------------------------------
__device__ __forceinline__ float block_reduce_sum(float v, float* red) {
  const int tid = threadIdx.x;
  red[tid] = v;
  __syncthreads();
  #pragma unroll
  for (int s = 128; s > 0; s >>= 1) {
    if (tid < s) red[tid] += red[tid + s];
    __syncthreads();
  }
  const float r = red[0];
  __syncthreads();
  return r;
}

// LN over 1024 cols + ELU, emit bf16 row at out + row*out_stride.
__global__ __launch_bounds__(256) void ln_elu_to_bf16(
    const float* __restrict__ X, const float* __restrict__ g,
    const float* __restrict__ be, bf16_t* __restrict__ out, int out_stride)
{
  __shared__ float red[256];
  const int row = blockIdx.x;
  const float* x = X + (size_t)row * 1024;
  float v[4];
  float s = 0.f;
  #pragma unroll
  for (int i = 0; i < 4; ++i) { v[i] = x[threadIdx.x + i * 256]; s += v[i]; }
  const float mean = block_reduce_sum(s, red) * (1.f / 1024.f);
  float s2 = 0.f;
  #pragma unroll
  for (int i = 0; i < 4; ++i) { const float d = v[i] - mean; s2 += d * d; }
  const float var  = block_reduce_sum(s2, red) * (1.f / 1024.f);
  const float rinv = rsqrtf(var + 1e-5f);
  #pragma unroll
  for (int i = 0; i < 4; ++i) {
    const int c = threadIdx.x + i * 256;
    float y = (v[i] - mean) * rinv * g[c] + be[c];
    y = y > 0.f ? y : expm1f(y);
    out[(size_t)row * out_stride + c] = (bf16_t)y;
  }
}

// LN over 3072 + GRU gates; updates deter (fp32 + bf16 + cat right half) and writes
// out[b][t][128:1152].
__global__ __launch_bounds__(256) void gru_update(
    const float* __restrict__ P, const float* __restrict__ g,
    const float* __restrict__ be, float* __restrict__ deter,
    bf16_t* __restrict__ cat, bf16_t* __restrict__ dbf,
    float* __restrict__ out, int t)
{
  __shared__ float red[256];
  const int row = blockIdx.x;
  const float* p = P + (size_t)row * 3072;
  float v[12];
  float s = 0.f;
  #pragma unroll
  for (int i = 0; i < 12; ++i) { v[i] = p[threadIdx.x + i * 256]; s += v[i]; }
  const float mean = block_reduce_sum(s, red) * (1.f / 3072.f);
  float s2 = 0.f;
  #pragma unroll
  for (int i = 0; i < 12; ++i) { const float d = v[i] - mean; s2 += d * d; }
  const float var  = block_reduce_sum(s2, red) * (1.f / 3072.f);
  const float rinv = rsqrtf(var + 1e-5f);
  #pragma unroll
  for (int j = 0; j < 4; ++j) {
    const int i = threadIdx.x + j * 256;
    const float p0 = (v[j]     - mean) * rinv * g[i]        + be[i];
    const float p1 = (v[j + 4] - mean) * rinv * g[1024 + i] + be[1024 + i];
    const float p2 = (v[j + 8] - mean) * rinv * g[2048 + i] + be[2048 + i];
    const float reset  = 1.f / (1.f + expf(-p0));
    const float cand   = tanhf(reset * p1);
    const float update = 1.f / (1.f + expf(-(p2 - 1.f)));   // UPDATE_BIAS = -1
    const size_t di = (size_t)row * 1024 + i;
    const float dn = update * cand + (1.f - update) * deter[di];
    deter[di] = dn;
    dbf[di] = (bf16_t)dn;
    cat[(size_t)row * 2048 + 1024 + i] = (bf16_t)dn;
    out[((size_t)row * 30 + t) * 1152 + 128 + i] = dn;
  }
}

// stats[B,128] -> mean/std, carry stoch, write out[b][t][0:128].
__global__ __launch_bounds__(256) void stats_out(
    const float* __restrict__ S, float* __restrict__ stoch,
    float* __restrict__ out, int t)
{
  const int idx = blockIdx.x * blockDim.x + threadIdx.x;  // B*64
  const int b = idx >> 6, j = idx & 63;
  const float mean = S[(size_t)b * 128 + j];
  const float sr   = S[(size_t)b * 128 + 64 + j];
  const float sp   = fmaxf(sr, 0.f) + log1pf(expf(-fabsf(sr))) + 0.1f;
  stoch[idx] = mean;
  const size_t o = ((size_t)b * 30 + t) * 1152;
  out[o + j]      = mean;
  out[o + 64 + j] = sp;
}

// concat(stoch, a_t) -> bf16 [B,96] (K padded 70 -> 96 with zeros)
__global__ __launch_bounds__(256) void build_img_in(
    const float* __restrict__ stoch, const float* __restrict__ action,
    bf16_t* __restrict__ out, int t)
{
  const int idx = blockIdx.x * blockDim.x + threadIdx.x;  // B*96
  const int b = idx / 96, c = idx % 96;
  float v = 0.f;
  if (c < 64)      v = stoch[b * 64 + c];
  else if (c < 70) v = action[((size_t)b * 30 + t) * 6 + (c - 64)];
  out[idx] = (bf16_t)v;
}

// W[K][N] fp32 -> Wt[N][Kpad] bf16 (zero-padded K)
__global__ __launch_bounds__(256) void transpose_to_bf16(
    const float* __restrict__ W, bf16_t* __restrict__ Wt, int K, int N, int Kpad)
{
  const int idx = blockIdx.x * blockDim.x + threadIdx.x;
  if (idx >= N * Kpad) return;
  const int n = idx / Kpad, k = idx % Kpad;
  const float v = (k < K) ? W[(size_t)k * N + n] : 0.f;
  Wt[idx] = (bf16_t)v;
}

__global__ __launch_bounds__(256) void init_state(
    float* __restrict__ deter, float* __restrict__ stoch,
    bf16_t* __restrict__ cat, bf16_t* __restrict__ dbf)
{
  const int idx = blockIdx.x * blockDim.x + threadIdx.x;  // 2048*1024
  deter[idx] = 0.f;
  dbf[idx] = (bf16_t)0.f;
  const int b = idx >> 10, i = idx & 1023;
  cat[(size_t)b * 2048 + 1024 + i] = (bf16_t)0.f;
  if (idx < 2048 * 64) stoch[idx] = 0.f;
}

// ---------------------------------------------------------------------------
extern "C" void kernel_launch(void* const* d_in, const int* in_sizes, int n_in,
                              void* d_out, int out_size, void* d_ws, size_t ws_size,
                              hipStream_t stream)
{
  const float* action   = (const float*)d_in[0];
  const float* W_img    = (const float*)d_in[1];
  const float* b_img    = (const float*)d_in[2];
  const float* g_img    = (const float*)d_in[3];
  const float* beta_img = (const float*)d_in[4];
  const float* W_gru    = (const float*)d_in[5];
  const float* b_gru    = (const float*)d_in[6];
  const float* g_gru    = (const float*)d_in[7];
  const float* beta_gru = (const float*)d_in[8];
  const float* W_e1     = (const float*)d_in[9];   // [5,1024,1024], use member 0
  const float* b_e1     = (const float*)d_in[10];  // [5,1024]
  const float* g_e1     = (const float*)d_in[11];
  const float* beta_e1  = (const float*)d_in[12];
  const float* W_e2     = (const float*)d_in[13];  // [5,1024,128], use member 0
  const float* b_e2     = (const float*)d_in[14];  // [5,128]
  float* out = (float*)d_out;

  char* ws = (char*)d_ws;
  size_t off = 0;
  auto alloc = [&](size_t bytes) -> char* {
    off = (off + 255) & ~(size_t)255;
    char* p = ws + off; off += bytes; return p;
  };
  bf16_t* wimg_t = (bf16_t*)alloc((size_t)1024 * 96 * 2);
  bf16_t* wgru_t = (bf16_t*)alloc((size_t)3072 * 2048 * 2);
  bf16_t* we1_t  = (bf16_t*)alloc((size_t)1024 * 1024 * 2);
  bf16_t* we2_t  = (bf16_t*)alloc((size_t)128 * 1024 * 2);
  bf16_t* img_in = (bf16_t*)alloc((size_t)2048 * 96 * 2);
  bf16_t* cat    = (bf16_t*)alloc((size_t)2048 * 2048 * 2);  // [x | deter] bf16
  bf16_t* dbf    = (bf16_t*)alloc((size_t)2048 * 1024 * 2);
  bf16_t* hbf    = (bf16_t*)alloc((size_t)2048 * 1024 * 2);
  float*  pre    = (float*)alloc((size_t)2048 * 3072 * 4);   // shared GEMM output
  float*  deter  = (float*)alloc((size_t)2048 * 1024 * 4);
  float*  stoch  = (float*)alloc((size_t)2048 * 64 * 4);
  float*  stats  = (float*)alloc((size_t)2048 * 128 * 4);

  // One-time weight prep (bf16 + transpose to [N][K])
  transpose_to_bf16<<<(1024 * 96 + 255) / 256, 256, 0, stream>>>(W_img, wimg_t, 70, 1024, 96);
  transpose_to_bf16<<<(3072 * 2048 + 255) / 256, 256, 0, stream>>>(W_gru, wgru_t, 2048, 3072, 2048);
  transpose_to_bf16<<<(1024 * 1024 + 255) / 256, 256, 0, stream>>>(W_e1, we1_t, 1024, 1024, 1024);
  transpose_to_bf16<<<(128 * 1024 + 255) / 256, 256, 0, stream>>>(W_e2, we2_t, 1024, 128, 1024);
  init_state<<<(2048 * 1024) / 256, 256, 0, stream>>>(deter, stoch, cat, dbf);

  for (int t = 0; t < 30; ++t) {
    // img_inp_layer
    build_img_in<<<(2048 * 96) / 256, 256, 0, stream>>>(stoch, action, img_in, t);
    gemm_bf16_wmma<<<dim3(1024 / 128, 2048 / 128), 256, 0, stream>>>(
        img_in, 96, wimg_t, 96, b_img, pre, 1024, 96);
    ln_elu_to_bf16<<<2048, 256, 0, stream>>>(pre, g_img, beta_img, cat, 2048);
    // GRU cell
    gemm_bf16_wmma<<<dim3(3072 / 128, 2048 / 128), 256, 0, stream>>>(
        cat, 2048, wgru_t, 2048, b_gru, pre, 3072, 2048);
    gru_update<<<2048, 256, 0, stream>>>(pre, g_gru, beta_gru, deter, cat, dbf, out, t);
    // ensemble head 0
    gemm_bf16_wmma<<<dim3(1024 / 128, 2048 / 128), 256, 0, stream>>>(
        dbf, 1024, we1_t, 1024, b_e1, pre, 1024, 1024);
    ln_elu_to_bf16<<<2048, 256, 0, stream>>>(pre, g_e1, beta_e1, hbf, 1024);
    gemm_bf16_wmma<<<dim3(128 / 128, 2048 / 128), 256, 0, stream>>>(
        hbf, 1024, we2_t, 1024, b_e2, stats, 128, 1024);
    stats_out<<<(2048 * 64) / 256, 256, 0, stream>>>(stats, stoch, out, t);
  }
  (void)in_sizes; (void)n_in; (void)out_size; (void)ws_size;
}
